// MultiModalModel_10471130267878
// MI455X (gfx1250) — compile-verified, matrix operand order checked by reading, not run on hardware
//
#include <hip/hip_runtime.h>

#define N_NODES  100000
#define N_EDGES  1600000
#define D        128
#define N_LAYERS 7
#define N_GRAPHS 64
#define N_CLASSES 10
#define SNN_IN   512
#define SNN_HID  1024
#define SNN_BETA 0.85f

typedef __attribute__((ext_vector_type(16))) _Float16 v16h;
typedef __attribute__((ext_vector_type(8)))  float    v8f;

// ---------------------------------------------------------------------------
// Pre-pack f32 weights into the per-lane v16h B-fragment layout for
// v_wmma_f32_16x16x32_f16.  B is 32x16 (KxN): lane n<16 holds column n,
// K = {kc+0..7, kc+16..23}; lane n+16 holds column n, K = {kc+8..15, kc+24..31}.
// fragId = ((layer*2 + mat)*8 + nTile)*4 + kc  -> 512 halves (32 lanes x 16).
// ---------------------------------------------------------------------------
__global__ void pack_weights_kernel(const float* __restrict__ wrel,
                                    const float* __restrict__ wroot,
                                    _Float16* __restrict__ wpack) {
    int fragId = blockIdx.x;           // 0..447
    int c = fragId & 3;                // kc chunk
    int t = (fragId >> 2) & 7;         // nTile
    int m = (fragId >> 5) & 1;         // 0 = Wrel, 1 = Wroot
    int l = fragId >> 6;               // layer
    const float* W = (m ? wroot : wrel) + (size_t)l * D * D;
    int lane = threadIdx.x;            // 0..31
    int col  = t * 16 + (lane & 15);
    int kb   = c * 32 + ((lane >> 4) << 3);
    _Float16* out = wpack + (size_t)fragId * 512 + lane * 16;
#pragma unroll
    for (int i = 0; i < 16; ++i) {
        int K = kb + (i < 8 ? i : i + 8);
        out[i] = (_Float16)W[K * D + col];
    }
}

__global__ void zero_kernel(float4* __restrict__ p, int n4) {
    int gid = blockIdx.x * blockDim.x + threadIdx.x;
    if (gid < n4) p[gid] = make_float4(0.f, 0.f, 0.f, 0.f);
}

// agg[dst] += h[src].  One wave per edge, 4 floats per lane.  The edge id is
// wave-uniform -> readfirstlane so the int64 index pair is fetched once per
// wave through the scalar path instead of 32x through VMEM.
__global__ void scatter_kernel(const float* __restrict__ h,
                               const long long* __restrict__ ei,
                               float* __restrict__ agg) {
    int gid = blockIdx.x * blockDim.x + threadIdx.x;
    int e = __builtin_amdgcn_readfirstlane(gid >> 5);
    int q = (threadIdx.x & 31) << 2;
    int s = (int)ei[e];
    int d = (int)ei[(size_t)N_EDGES + e];
    float4 v = *(const float4*)(h + (size_t)s * D + q);
    float* a = agg + (size_t)d * D + q;
    atomicAdd(a + 0, v.x);
    atomicAdd(a + 1, v.y);
    atomicAdd(a + 2, v.z);
    atomicAdd(a + 3, v.w);
}

// Load a 16x32 f16 A-fragment from a row-major f32 row pointer.
// Lane layout: row = lane&15, K = kb + {0..7, 16..23} (kb includes the +8 for
// lanes 16..31).  Two contiguous 8-float runs -> four b128 loads.
__device__ __forceinline__ v16h load_a_frag(const float* __restrict__ row, int kb) {
    const float4* p0 = (const float4*)(row + kb);
    const float4* p1 = (const float4*)(row + kb + 16);
    float4 x0 = p0[0], x1 = p0[1], y0 = p1[0], y1 = p1[1];
    v16h a;
    a[0]  = (_Float16)x0.x; a[1]  = (_Float16)x0.y;
    a[2]  = (_Float16)x0.z; a[3]  = (_Float16)x0.w;
    a[4]  = (_Float16)x1.x; a[5]  = (_Float16)x1.y;
    a[6]  = (_Float16)x1.z; a[7]  = (_Float16)x1.w;
    a[8]  = (_Float16)y0.x; a[9]  = (_Float16)y0.y;
    a[10] = (_Float16)y0.z; a[11] = (_Float16)y0.w;
    a[12] = (_Float16)y1.x; a[13] = (_Float16)y1.y;
    a[14] = (_Float16)y1.z; a[15] = (_Float16)y1.w;
    return a;
}

// out = relu(agg @ Wrel + brel + h @ Wroot).
// One 16-row block per CTA; 8 waves.  Stage: wave w converts exactly one of
// the 8 A-fragments (mat = w>>2, kc = w&3) f32->f16 into LDS (8 KB), so the
// unique 16 KB of A data is read from global and converted exactly once per
// block.  Compute: wave w owns N-tile w; 4 K-steps x 2 matrices = 8 WMMAs.
__global__ void gemm_layer_kernel(const float* __restrict__ agg,
                                  const float* __restrict__ h,
                                  const _Float16* __restrict__ wpack,
                                  const float* __restrict__ brel,
                                  float* __restrict__ out,
                                  int layer) {
    __shared__ v16h sA[2][4][32];      // [mat][kc][lane] = 8 KB

    int lane  = threadIdx.x & 31;
    int wv    = threadIdx.x >> 5;      // wave 0..7
    int mTile = blockIdx.x;            // 0..6249
    int koff  = (lane >> 4) << 3;      // +8 K offset for lanes 16..31

    {   // ---- stage A fragments through LDS ----
        int m = wv >> 2;               // 0 = agg, 1 = h
        int c = wv & 3;                // kc chunk
        const float* base = (m ? h : agg) + (size_t)(mTile * 16 + (lane & 15)) * D;
        sA[m][c][lane] = load_a_frag(base, c * 32 + koff);
    }
    __syncthreads();

    const v16h* bp = (const v16h*)wpack;
    int fbRel  = ((layer * 2 + 0) * 8 + wv) * 4;
    int fbRoot = ((layer * 2 + 1) * 8 + wv) * 4;

    v8f acc = {};
#pragma unroll
    for (int c = 0; c < 4; ++c) {
        v16h aA = sA[0][c][lane];
        v16h aH = sA[1][c][lane];
        v16h bA = bp[(size_t)(fbRel  + c) * 32 + lane];
        v16h bH = bp[(size_t)(fbRoot + c) * 32 + lane];
        acc = __builtin_amdgcn_wmma_f32_16x16x32_f16(false, aA, false, bA,
                                                     (short)0, acc, false, false);
        acc = __builtin_amdgcn_wmma_f32_16x16x32_f16(false, aH, false, bH,
                                                     (short)0, acc, false, false);
    }

    int col  = wv * 16 + (lane & 15);
    int orow = mTile * 16 + ((lane >> 4) << 3);   // + v gives matrix row
    float b  = brel[col];
#pragma unroll
    for (int v = 0; v < 8; ++v) {
        float val = acc[v] + b;
        out[(size_t)(orow + v) * D + col] = fmaxf(val, 0.f);
    }
}

// pooled[batch[n]] += h[n].  One wave per node; node id is wave-uniform.
__global__ void pool_accum_kernel(const float* __restrict__ h,
                                  const long long* __restrict__ batch,
                                  float* __restrict__ pooled) {
    int gid = blockIdx.x * blockDim.x + threadIdx.x;
    int n = __builtin_amdgcn_readfirstlane(gid >> 5);
    int q = (threadIdx.x & 31) << 2;
    int g = (int)batch[n];
    float4 v = *(const float4*)(h + (size_t)n * D + q);
    float* a = pooled + (size_t)g * D + q;
    atomicAdd(a + 0, v.x);
    atomicAdd(a + 1, v.y);
    atomicAdd(a + 2, v.z);
    atomicAdd(a + 3, v.w);
}

__global__ void count_kernel(const long long* __restrict__ batch,
                             float* __restrict__ cnt) {
    int n = blockIdx.x * blockDim.x + threadIdx.x;
    if (n < N_NODES) atomicAdd(cnt + (int)batch[n], 1.f);
}

__global__ void gnn_head_kernel(const float* __restrict__ pooled,
                                const float* __restrict__ cnt,
                                const float* __restrict__ lin_w,
                                const float* __restrict__ lin_b,
                                float* __restrict__ gnn_logits) {
    int tid = blockIdx.x * blockDim.x + threadIdx.x;
    if (tid >= N_GRAPHS * N_CLASSES) return;
    int g = tid / N_CLASSES, c = tid % N_CLASSES;
    float inv = 1.f / fmaxf(cnt[g], 1.f);
    float s = lin_b[c];
    for (int k = 0; k < D; ++k)
        s += pooled[g * D + k] * inv * lin_w[k * N_CLASSES + c];
    gnn_logits[tid] = s;
}

__global__ void snn1_kernel(const float* __restrict__ x,
                            const float* __restrict__ w1,
                            const float* __restrict__ b1,
                            float* __restrict__ hh) {
    int tid = blockIdx.x * blockDim.x + threadIdx.x;   // 64*1024
    int g = tid >> 10, j = tid & 1023;
    float s = b1[j];
    for (int k = 0; k < SNN_IN; ++k)
        s += x[g * SNN_IN + k] * w1[k * SNN_HID + j];
    hh[tid] = fmaxf(s, 0.f);
}

__global__ void snn2_kernel(const float* __restrict__ hh,
                            const float* __restrict__ w2,
                            const float* __restrict__ b2,
                            float* __restrict__ snn_logits) {
    int tid = blockIdx.x * blockDim.x + threadIdx.x;
    if (tid >= N_GRAPHS * N_CLASSES) return;
    int g = tid / N_CLASSES, c = tid % N_CLASSES;
    float s = b2[c];
    for (int k = 0; k < SNN_HID; ++k)
        s += hh[g * SNN_HID + k] * w2[k * N_CLASSES + c];
    snn_logits[tid] = SNN_BETA * s;
}

__global__ void fuse_kernel(const float* __restrict__ snn_logits,
                            const float* __restrict__ gnn_logits,
                            const float* __restrict__ fw,
                            const float* __restrict__ fb,
                            float* __restrict__ out) {
    int tid = blockIdx.x * blockDim.x + threadIdx.x;
    if (tid >= N_GRAPHS * N_CLASSES) return;
    int g = tid / N_CLASSES, c = tid % N_CLASSES;
    float s = fb[c];
    for (int j = 0; j < N_CLASSES; ++j) {
        s += snn_logits[g * N_CLASSES + j] * fw[j * N_CLASSES + c];
        s += gnn_logits[g * N_CLASSES + j] * fw[(N_CLASSES + j) * N_CLASSES + c];
    }
    out[tid] = s;
}

static inline size_t align_up(size_t v, size_t a) { return (v + a - 1) & ~(a - 1); }

extern "C" void kernel_launch(void* const* d_in, const int* in_sizes, int n_in,
                              void* d_out, int out_size, void* d_ws, size_t ws_size,
                              hipStream_t stream) {
    const float*     snn_x      = (const float*)d_in[0];
    const float*     x          = (const float*)d_in[1];
    const long long* edge_index = (const long long*)d_in[2];
    const long long* batch      = (const long long*)d_in[3];
    const float*     snn_w1     = (const float*)d_in[4];
    const float*     snn_b1     = (const float*)d_in[5];
    const float*     snn_w2     = (const float*)d_in[6];
    const float*     snn_b2     = (const float*)d_in[7];
    const float*     conv_wrel  = (const float*)d_in[8];
    const float*     conv_wroot = (const float*)d_in[9];
    const float*     conv_brel  = (const float*)d_in[10];
    const float*     lin_w      = (const float*)d_in[11];
    const float*     lin_b      = (const float*)d_in[12];
    const float*     fuse_w     = (const float*)d_in[13];
    const float*     fuse_b     = (const float*)d_in[14];
    float* out = (float*)d_out;

    // ---- workspace carve-up ----
    char* p = (char*)d_ws;
    size_t off = 0;
    const size_t hbytes = (size_t)N_NODES * D * sizeof(float);   // 51.2 MB
    float* h0  = (float*)(p + off); off = align_up(off + hbytes, 256);
    float* h1  = (float*)(p + off); off = align_up(off + hbytes, 256);
    float* agg = (float*)(p + off); off = align_up(off + hbytes, 256);
    _Float16* wpack = (_Float16*)(p + off);
    off = align_up(off + (size_t)N_LAYERS * 2 * 8 * 4 * 512 * sizeof(_Float16), 256);
    float* pooled = (float*)(p + off); off = align_up(off + N_GRAPHS * D * sizeof(float), 256);
    float* cnt    = (float*)(p + off); off = align_up(off + N_GRAPHS * sizeof(float), 256);
    float* gnnlog = (float*)(p + off); off = align_up(off + N_GRAPHS * N_CLASSES * sizeof(float), 256);
    float* snnh   = (float*)(p + off); off = align_up(off + (size_t)N_GRAPHS * SNN_HID * sizeof(float), 256);
    float* snnlog = (float*)(p + off); off = align_up(off + N_GRAPHS * N_CLASSES * sizeof(float), 256);
    (void)ws_size; (void)in_sizes; (void)n_in; (void)out_size;

    // ---- pack weight fragments (once per call) ----
    pack_weights_kernel<<<N_LAYERS * 2 * 8 * 4, 32, 0, stream>>>(conv_wrel, conv_wroot, wpack);

    // ---- GNN trunk ----
    const int aggN4 = N_NODES * D / 4;                 // 3.2M float4
    float* hbuf[2] = { h0, h1 };
    const float* hprev = x;
    for (int l = 0; l < N_LAYERS; ++l) {
        zero_kernel<<<(aggN4 + 255) / 256, 256, 0, stream>>>((float4*)agg, aggN4);
        scatter_kernel<<<(N_EDGES * 32) / 256, 256, 0, stream>>>(hprev, edge_index, agg);
        float* hnext = hbuf[l & 1];
        gemm_layer_kernel<<<N_NODES / 16, 256, 0, stream>>>(agg, hprev, wpack,
                                                            conv_brel + l * D, hnext, l);
        hprev = hnext;
    }

    // ---- pooling + GNN head ----
    zero_kernel<<<(N_GRAPHS * D / 4 + 255) / 256, 256, 0, stream>>>((float4*)pooled, N_GRAPHS * D / 4);
    zero_kernel<<<1, 256, 0, stream>>>((float4*)cnt, N_GRAPHS / 4);
    pool_accum_kernel<<<(N_NODES * 32) / 256, 256, 0, stream>>>(hprev, batch, pooled);
    count_kernel<<<(N_NODES + 255) / 256, 256, 0, stream>>>(batch, cnt);
    gnn_head_kernel<<<(N_GRAPHS * N_CLASSES + 255) / 256, 256, 0, stream>>>(pooled, cnt,
                                                                            lin_w, lin_b, gnnlog);

    // ---- SNN branch ----
    snn1_kernel<<<(N_GRAPHS * SNN_HID) / 256, 256, 0, stream>>>(snn_x, snn_w1, snn_b1, snnh);
    snn2_kernel<<<(N_GRAPHS * N_CLASSES + 255) / 256, 256, 0, stream>>>(snnh, snn_w2, snn_b2, snnlog);

    // ---- fusion ----
    fuse_kernel<<<(N_GRAPHS * N_CLASSES + 255) / 256, 256, 0, stream>>>(snnlog, gnnlog,
                                                                        fuse_w, fuse_b, out);
}
